// GraphAttnApprox_63513976373281
// MI455X (gfx1250) — matrix operand order repaired and checked by reading, not compiled, not run on hardware
//
#include <hip/hip_runtime.h>
#include <hip/hip_bf16.h>

typedef unsigned int u32;
typedef _Float16 half2_t __attribute__((ext_vector_type(2)));
typedef _Float16 v16h    __attribute__((ext_vector_type(16)));
typedef float    v8f     __attribute__((ext_vector_type(8)));
typedef int      v4i_t   __attribute__((vector_size(16)));

struct __attribute__((aligned(16))) U4 { u32 x, y, z, w; };
struct __attribute__((aligned(16))) F4 { float x, y, z, w; };

union Frag {
    v16h    h;
    half2_t p[8];
    u32     u[8];
    U4      q[2];
};

static __device__ __forceinline__ half2_t pk2(float a, float b) {
    return (half2_t)__builtin_amdgcn_cvt_pkrtz(a, b);
}

#define HAVE_ASYNC_LDS                                                     \
    (__has_builtin(__builtin_amdgcn_global_load_async_to_lds_b128) &&      \
     __has_builtin(__builtin_amdgcn_s_wait_asynccnt))

// ---------------------------------------------------------------------------
// Kernel 0: pre-pack W2 [64][128] f32 -> packed f16 pairs [64][64] dwords.
// ---------------------------------------------------------------------------
__global__ void pack_w2_kernel(const float* __restrict__ W2,
                               u32* __restrict__ w2pk) {
    int idx = blockIdx.x * 256 + threadIdx.x;      // 0..4095
    int nn = idx >> 6, kp = idx & 63;
    half2_t pk = pk2(W2[nn * 128 + kp * 2], W2[nn * 128 + kp * 2 + 1]);
    w2pk[idx] = __builtin_bit_cast(u32, pk);
}

// ---------------------------------------------------------------------------
// Kernel 1: LayerNorm, output packed f16 (xn). One block per row of 2048.
// ---------------------------------------------------------------------------
__global__ void ln_kernel(const float* __restrict__ x,
                          const float* __restrict__ g,
                          const float* __restrict__ b,
                          _Float16* __restrict__ xnpk) {
    const int row = blockIdx.x;
    const int t = threadIdx.x;              // 256 threads
    const float* xr = x + row * 2048;

    float vals[8];
    float s = 0.f, ss = 0.f;
#pragma unroll
    for (int k = 0; k < 8; ++k) {
        float v = xr[t + 256 * k];
        vals[k] = v;
        s += v; ss += v * v;
    }
#pragma unroll
    for (int off = 16; off >= 1; off >>= 1) {
        s  += __shfl_xor(s,  off, 32);
        ss += __shfl_xor(ss, off, 32);
    }
    __shared__ float ws1[8], ws2[8];
    __shared__ float smu, srs;
    int wave = t >> 5, lane = t & 31;
    if (lane == 0) { ws1[wave] = s; ws2[wave] = ss; }
    __syncthreads();
    if (t == 0) {
        float S = 0.f, SS = 0.f;
        for (int w = 0; w < 8; ++w) { S += ws1[w]; SS += ws2[w]; }
        float mu  = S * (1.0f / 2048.0f);
        float var = SS * (1.0f / 2048.0f) - mu * mu;
        smu = mu;
        srs = rsqrtf(var + 1e-5f);
    }
    __syncthreads();
    float mu = smu, rs = srs;
#pragma unroll
    for (int k = 0; k < 8; ++k) {
        int c = t + 256 * k;
        float xn = (vals[k] - mu) * rs * g[c] + b[c];
        xnpk[row * 2048 + c] = (_Float16)xn;
    }
}

// ---------------------------------------------------------------------------
// Kernel 2/6: generic WMMA GEMM  C[M,N] = A(f16 packed, row-major) * W[N,K]^T
// Block = 128 threads = 4 waves. wave -> (mtile = blockIdx.x,
// ntile = blockIdx.y*4 + wave). Kdim multiple of 32.
// Epilogue: optional bias[n], optional residual[row*N+n]; store f32 or f16.
// ---------------------------------------------------------------------------
__global__ void gemm_wmma_kernel(const u32* __restrict__ Apk, int astride_dw,
                                 const float* __restrict__ W, int Kdim,
                                 const float* __restrict__ bias,
                                 const float* __restrict__ resid,
                                 float* __restrict__ Cf,
                                 _Float16* __restrict__ Ch, int N) {
    const int wave = threadIdx.x >> 5;
    const int lane = threadIdx.x & 31;
    const int mtile = blockIdx.x;
    const int ntile = blockIdx.y * 4 + wave;
    const int m  = lane & 15;
    const int hi = lane >> 4;
    const int arow = mtile * 16 + m;     // A row this lane supplies
    const int n    = ntile * 16 + m;     // W row / C column this lane supplies

    v8f acc = {0.f, 0.f, 0.f, 0.f, 0.f, 0.f, 0.f, 0.f};
    const int nchunks = Kdim >> 5;

    for (int kc = 0; kc < nchunks; ++kc) {
        // A fragment: 16-bit A layout; dwords 0-3 = kpairs base..base+3,
        // dwords 4-7 = kpairs base+8..base+11, base = kc*16 + hi*4.
        Frag a;
        const u32* ap = Apk + arow * astride_dw + kc * 16 + hi * 4;
        a.q[0] = *(const U4*)ap;
        a.q[1] = *(const U4*)(ap + 8);

        // B fragment: lane holds column n; K = kc*32 + hi*8 + [0..7], kc*32 + 16 + hi*8 + [0..7].
        Frag bf;
        const float* wr = W + n * Kdim + kc * 32 + hi * 8;
        F4 w0 = *(const F4*)(wr);
        F4 w1 = *(const F4*)(wr + 4);
        F4 w2 = *(const F4*)(wr + 16);
        F4 w3 = *(const F4*)(wr + 20);
        bf.p[0] = pk2(w0.x, w0.y); bf.p[1] = pk2(w0.z, w0.w);
        bf.p[2] = pk2(w1.x, w1.y); bf.p[3] = pk2(w1.z, w1.w);
        bf.p[4] = pk2(w2.x, w2.y); bf.p[5] = pk2(w2.z, w2.w);
        bf.p[6] = pk2(w3.x, w3.y); bf.p[7] = pk2(w3.z, w3.w);

        acc = __builtin_amdgcn_wmma_f32_16x16x32_f16(false, a.h, false, bf.h,
                                                     (short)0, acc, false, false);
    }

    const float bv = bias ? bias[n] : 0.0f;
#pragma unroll
    for (int r = 0; r < 8; ++r) {
        int row = mtile * 16 + r + hi * 8;   // C layout: M = r + 8*hi
        float c = acc[r] + bv;
        if (resid) c += resid[row * N + n];
        if (Cf) Cf[row * N + n] = c;
        else    Ch[row * N + n] = (_Float16)c;
    }
}

// ---------------------------------------------------------------------------
// Kernel 3: pairwise graph MLP -> sim[i][j].
// grid = (1024 i, 4 jblocks), block = 128 threads (4 waves).
// Each wave iterates 4 M-tiles of 16 pairs; per tile: K=128 (4 chunks),
// N=64 (4 n-tiles) -> 16 WMMAs.
// W2 arrives pre-packed (f16 pairs); staged global->LDS with the CDNA5
// async-to-LDS path (ASYNCcnt) when available.
// ---------------------------------------------------------------------------
__global__ void pair_sim_kernel(const u32* __restrict__ hpk,   // [1024][64] dw (f16x2)
                                const u32* __restrict__ w2pk,  // [64][64] dw (f16x2)
                                const float* __restrict__ b1,  // [128]
                                const float* __restrict__ b2,  // [64]
                                const float* __restrict__ W3,  // [64]
                                const float* __restrict__ b3,  // [1]
                                float* __restrict__ sim) {
    __shared__ u32 s_w2[64 * 64];   // [n][kpair] packed f16 pairs (B-frag order)
    __shared__ u32 s_hib[64];       // h[i] + b1, packed f16 pairs

    const int tid = threadIdx.x;
    const int i = blockIdx.x;

#if HAVE_ASYNC_LDS
    // 4096 dwords = 1024 x b128 transfers; 8 per thread.
#pragma unroll
    for (int it = 0; it < 8; ++it) {
        int idx4 = (tid + it * 128) * 4;   // dword index, 16B aligned
        __builtin_amdgcn_global_load_async_to_lds_b128(
            (__attribute__((address_space(1))) v4i_t*)(w2pk + idx4),
            (__attribute__((address_space(3))) v4i_t*)(s_w2 + idx4),
            0, 0);
    }
#else
    for (int idx = tid; idx < 64 * 64; idx += 128)
        s_w2[idx] = w2pk[idx];
#endif
    // Stage hib = h[i] + b1 (64 dwords)
    if (tid < 64) {
        half2_t hv = __builtin_bit_cast(half2_t, hpk[i * 64 + tid]);
        half2_t bb = pk2(b1[tid * 2], b1[tid * 2 + 1]);
        half2_t sm = hv + bb;
        s_hib[tid] = __builtin_bit_cast(u32, sm);
    }
#if HAVE_ASYNC_LDS
    __builtin_amdgcn_s_wait_asynccnt(0);
#endif
    __syncthreads();

    const int wave = tid >> 5;
    const int lane = tid & 31;
    const int m  = lane & 15;
    const int hi = lane >> 4;

    float b2v[4], w3v[4];
#pragma unroll
    for (int nt = 0; nt < 4; ++nt) {
        b2v[nt] = b2[nt * 16 + m];
        w3v[nt] = W3[nt * 16 + m];
    }
    const float bb3 = b3[0];
    const half2_t z = {(_Float16)0.0f, (_Float16)0.0f};

    for (int it = 0; it < 4; ++it) {
        const int j0 = blockIdx.y * 256 + (it * 4 + wave) * 16;
        const int j  = j0 + m;           // this lane's pair row (M = m)

        v8f acc[4];
#pragma unroll
        for (int nt = 0; nt < 4; ++nt)
            acc[nt] = (v8f){0.f,0.f,0.f,0.f,0.f,0.f,0.f,0.f};

#pragma unroll
        for (int kc = 0; kc < 4; ++kc) {
            // A fragment: f = relu(hib - h_j), built in 16-bit A layout.
            Frag hj, hb, a;
            const u32* hp = hpk + j * 64 + kc * 16 + hi * 4;
            hj.q[0] = *(const U4*)hp;
            hj.q[1] = *(const U4*)(hp + 8);
            const u32* hbp = s_hib + kc * 16 + hi * 4;
            hb.q[0] = *(const U4*)hbp;
            hb.q[1] = *(const U4*)(hbp + 8);
#pragma unroll
            for (int d = 0; d < 8; ++d) {
                half2_t diff = hb.p[d] - hj.p[d];
                a.p[d] = __builtin_elementwise_max(diff, z);
            }
#pragma unroll
            for (int nt = 0; nt < 4; ++nt) {
                Frag bf;
                const u32* wp = s_w2 + (nt * 16 + m) * 64 + kc * 16 + hi * 4;
                bf.q[0] = *(const U4*)wp;
                bf.q[1] = *(const U4*)(wp + 8);
                acc[nt] = __builtin_amdgcn_wmma_f32_16x16x32_f16(
                    false, a.h, false, bf.h, (short)0, acc[nt], false, false);
            }
        }

        // Epilogue: relu(acc + b2) dot W3, reduce over the 16 N-lanes.
        float part[8];
#pragma unroll
        for (int r = 0; r < 8; ++r) {
            float sv = 0.f;
#pragma unroll
            for (int nt = 0; nt < 4; ++nt)
                sv += fmaxf(acc[nt][r] + b2v[nt], 0.f) * w3v[nt];
            part[r] = sv;
        }
#pragma unroll
        for (int r = 0; r < 8; ++r) {
#pragma unroll
            for (int off = 1; off < 16; off <<= 1)
                part[r] += __shfl_xor(part[r], off, 32);
        }
#pragma unroll
        for (int r = 0; r < 8; ++r) {
            if (m == r) {                       // two writers: hi=0 -> M=r, hi=1 -> M=8+r
                int jg = j0 + r + hi * 8;
                sim[i * 1024 + jg] = (jg == i) ? 0.0f : (part[r] + bb3);
            }
        }
    }
}

// ---------------------------------------------------------------------------
// Kernel 4: per-row threshold = 102nd largest sim value (radix select).
// ---------------------------------------------------------------------------
__global__ void topk_thresh_kernel(const float* __restrict__ sim,
                                   float* __restrict__ thresh) {
    __shared__ int cnt[256];
    __shared__ unsigned sh_prefix;
    __shared__ int sh_k;
    const int i = blockIdx.x, t = threadIdx.x;
    if (t == 0) { sh_prefix = 0u; sh_k = 102; }

    for (int p = 0; p < 4; ++p) {
        cnt[t] = 0;
        __syncthreads();
        const unsigned pref = sh_prefix;
        const int shift = 24 - p * 8;
        const unsigned himask = (p == 0) ? 0u : (0xFFFFFFFFu << (shift + 8));
#pragma unroll
        for (int k2 = 0; k2 < 4; ++k2) {
            float f = sim[i * 1024 + t + k2 * 256];
            unsigned fb = __float_as_uint(f);
            unsigned u = (fb & 0x80000000u) ? ~fb : (fb | 0x80000000u);
            if ((u & himask) == (pref & himask))
                atomicAdd(&cnt[(u >> shift) & 255], 1);
        }
        __syncthreads();
        if (t == 0) {
            int kk = sh_k, run = 0, bsel = 0;
            for (int bq = 255; bq >= 0; --bq) {
                int c = cnt[bq];
                if (run + c >= kk) { bsel = bq; sh_k = kk - run; break; }
                run += c;
            }
            sh_prefix = sh_prefix | ((unsigned)bsel << shift);
        }
        __syncthreads();
    }
    if (t == 0) {
        unsigned u = sh_prefix;
        unsigned fb = (u & 0x80000000u) ? (u ^ 0x80000000u) : ~u;
        thresh[i] = __uint_as_float(fb);
    }
}

// ---------------------------------------------------------------------------
// Kernel 5: masked attention (32 heads x dim 2). Block per query row i,
// 256 threads: lane -> head (t&31), partition (t>>5) covers 128 keys.
// Online softmax per partition; merge partials via LDS. ao stored packed f16.
// ---------------------------------------------------------------------------
__global__ void attn_kernel(const float* __restrict__ qkv,
                            const float* __restrict__ sim,
                            const float* __restrict__ thresh,
                            u32* __restrict__ aopk) {
    struct P { float m, l, a0, a1; };
    __shared__ P red[8][32];
    const int i = blockIdx.x, t = threadIdx.x;
    const int h = t & 31, part = t >> 5;
    const float th = thresh[i];
    const float q0 = qkv[i * 192 + 2 * h];
    const float q1 = qkv[i * 192 + 2 * h + 1];

    float m = -INFINITY, l = 0.f, a0 = 0.f, a1 = 0.f;
    const float* simr = sim + i * 1024;
    for (int j = part * 128; j < part * 128 + 128; ++j) {
        float sv = simr[j];
        if (sv >= th) {
            float k0 = qkv[j * 192 + 64 + 2 * h];
            float k1 = qkv[j * 192 + 65 + 2 * h];
            float sc = (q0 * k0 + q1 * k1) * 0.70710678118654752f;
            float nm = fmaxf(m, sc);
            float w  = __expf(m - nm);       // m = -inf -> 0
            float pw = __expf(sc - nm);
            float v0 = qkv[j * 192 + 128 + 2 * h];
            float v1 = qkv[j * 192 + 129 + 2 * h];
            l  = l * w + pw;
            a0 = a0 * w + pw * v0;
            a1 = a1 * w + pw * v1;
            m = nm;
        }
    }
    red[part][h] = P{m, l, a0, a1};
    __syncthreads();

    if (t < 32) {
        float M = -INFINITY, L = 0.f, A0 = 0.f, A1 = 0.f;
#pragma unroll
        for (int p2 = 0; p2 < 8; ++p2) {
            P r = red[p2][t];
            float nm = fmaxf(M, r.m);
            if (nm != -INFINITY) {
                float wa = (M   == -INFINITY) ? 0.f : __expf(M - nm);
                float wb = (r.m == -INFINITY) ? 0.f : __expf(r.m - nm);
                L  = L  * wa + r.l  * wb;
                A0 = A0 * wa + r.a0 * wb;
                A1 = A1 * wa + r.a1 * wb;
                M = nm;
            }
        }
        float inv = 1.0f / L;
        half2_t o = pk2(A0 * inv, A1 * inv);
        aopk[i * 32 + t] = __builtin_bit_cast(u32, o);
    }
}

// ---------------------------------------------------------------------------
// Launcher
// ---------------------------------------------------------------------------
extern "C" void kernel_launch(void* const* d_in, const int* in_sizes, int n_in,
                              void* d_out, int out_size, void* d_ws, size_t ws_size,
                              hipStream_t stream) {
    (void)in_sizes; (void)n_in; (void)out_size; (void)ws_size;
    const float* x     = (const float*)d_in[0];
    const float* gamma = (const float*)d_in[1];
    const float* beta  = (const float*)d_in[2];
    const float* W1    = (const float*)d_in[3];
    const float* b1    = (const float*)d_in[4];
    const float* W2    = (const float*)d_in[5];
    const float* b2    = (const float*)d_in[6];
    const float* W3    = (const float*)d_in[7];
    const float* b3    = (const float*)d_in[8];
    const float* Wqkv  = (const float*)d_in[9];
    const float* bqkv  = (const float*)d_in[10];
    const float* Wout  = (const float*)d_in[11];
    const float* bout  = (const float*)d_in[12];

    char* w = (char*)d_ws;
    _Float16* xnpk = (_Float16*)(w);                              // 4 MiB
    u32*      hpk  = (u32*)(w + (4u << 20));                      // 256 KiB
    float*    qkv  = (float*)(w + (4u << 20) + (256u << 10));     // 768 KiB
    float*    sim  = (float*)(w + (5u << 20));                    // 4 MiB
    float*    thr  = (float*)(w + (9u << 20));                    // 4 KiB
    u32*      aopk = (u32*)(w + (9u << 20) + (64u << 10));        // 128 KiB
    u32*      w2pk = (u32*)(w + (10u << 20));                     // 16 KiB

    // 0) pre-pack W2 to f16 pairs (B-fragment order)
    pack_w2_kernel<<<16, 256, 0, stream>>>(W2, w2pk);

    // 1) LayerNorm -> xn (f16 packed)
    ln_kernel<<<1024, 256, 0, stream>>>(x, gamma, beta, xnpk);

    // 2a) h = xn @ W1^T  (M=1024, K=2048, N=128) -> f16 packed
    gemm_wmma_kernel<<<dim3(64, 2), 128, 0, stream>>>(
        (const u32*)xnpk, 1024, W1, 2048, nullptr, nullptr,
        nullptr, (_Float16*)hpk, 128);

    // 2b) qkv = xn @ Wqkv^T + bqkv  (N=192) -> f32
    gemm_wmma_kernel<<<dim3(64, 3), 128, 0, stream>>>(
        (const u32*)xnpk, 1024, Wqkv, 2048, bqkv, nullptr,
        qkv, nullptr, 192);

    // 3) pairwise MLP -> sim[1024][1024]
    pair_sim_kernel<<<dim3(1024, 4), 128, 0, stream>>>(
        hpk, w2pk, b1, b2, W3, b3, sim);

    // 4) top-k (k=102) threshold per row
    topk_thresh_kernel<<<1024, 256, 0, stream>>>(sim, thr);

    // 5) masked attention -> ao (f16 packed)
    attn_kernel<<<1024, 256, 0, stream>>>(qkv, sim, thr, aopk);

    // 6) out = x + ao @ Wout^T + bout  (M=1024, K=64, N=2048) -> f32
    gemm_wmma_kernel<<<dim3(64, 32), 128, 0, stream>>>(
        aopk, 32, Wout, 64, bout, x, (float*)d_out, nullptr, 2048);
}
